// BuNN_10797547782301
// MI455X (gfx1250) — matrix operand order repaired and checked by reading, not compiled, not run on hardware
//
#include <hip/hip_runtime.h>

// ---------------------------------------------------------------------------
// BuNN forward for MI455X (gfx1250).
// Dominant cost: 16x (L @ term) with L 8192x8192. Strategy:
//   - convert L to bf16 once (128MB -> L2-resident, streamed 16x)
//   - term/result transposed (64 x 8192): WMMA B-fragments and C/D epilogue
//     stores are fully contiguous
//   - v_wmma_f32_16x16x32_bf16, f32 accumulation
//   - K-split x8 (4096 waves) + 2-stage software pipeline so loads for the
//     next K-chunk are in flight while WMMAs consume the current one
// ---------------------------------------------------------------------------

typedef __attribute__((ext_vector_type(16))) __bf16 v16bf;
typedef __attribute__((ext_vector_type(8)))  __bf16 v8bf;
typedef __attribute__((ext_vector_type(4)))  __bf16 v4bf;
typedef __attribute__((ext_vector_type(8)))  float  v8f;

#define NROWS  8192
#define D_IN   128
#define TOT    64      // 2*B
#define NBUN   32      // bundles
#define KDIM   8192    // diffusion inner dim (== NROWS)
#define KSPLIT 8
#define KCHUNK (KDIM / KSPLIT)   // 1024

__device__ __forceinline__ float gelu_exact(float x) {
  return 0.5f * x * (1.0f + erff(x * 0.70710678118654752440f));
}

#define WMMA_BF16(A, B, C) \
  __builtin_amdgcn_wmma_f32_16x16x32_bf16(false, (A), false, (B), (short)0, (C), false, false)

// --------------------------- f32 -> bf16 convert ---------------------------
__global__ void cvt_bf16_kernel(const float* __restrict__ src,
                                __bf16* __restrict__ dst, long n) {
  long i = ((long)blockIdx.x * blockDim.x + threadIdx.x) * 4;
  long stride = (long)gridDim.x * blockDim.x * 4;
  for (; i < n; i += stride) {
    float4 v = *(const float4*)(src + i);
    v4bf o;
    o[0] = (__bf16)v.x; o[1] = (__bf16)v.y;
    o[2] = (__bf16)v.z; o[3] = (__bf16)v.w;
    *(v4bf*)(dst + i) = o;
  }
}

// ------------------- small dense GEMM: C = act(A W^T + b) ------------------
__global__ void gemm_act_kernel(const float* __restrict__ A,
                                const float* __restrict__ W,
                                const float* __restrict__ bias,
                                float* __restrict__ C,
                                int nrows, int din, int dout, int act) {
  int g = blockIdx.x * blockDim.x + threadIdx.x;
  if (g >= nrows * dout) return;
  int row = g / dout, col = g % dout;
  const float* a = A + (size_t)row * din;
  const float* w = W + (size_t)col * din;
  float s = bias[col];
  #pragma unroll 8
  for (int k = 0; k < din; ++k) s = fmaf(a[k], w[k], s);
  if (act) s = gelu_exact(s);
  C[g] = s;
}

// -------- rotate(transpose=True) + H = hb @ lt_w^T + lt_b ------------------
__global__ __launch_bounds__(256) void rotate_lt_kernel(
    const float* __restrict__ h, const float* __restrict__ ang,
    const float* __restrict__ ltw, const float* __restrict__ ltb,
    float* __restrict__ resultT, __bf16* __restrict__ termT) {
  __shared__ float hb[4][TOT];
  int t = threadIdx.x;
  int row0 = blockIdx.x * 4;
  if (t < 128) {                       // 4 rows x 32 bundles
    int rl = t >> 5, b = t & 31;
    int row = row0 + rl;
    float a = ang[(size_t)row * NBUN + b];
    float c = cosf(a), s = sinf(a);
    float xc = h[(size_t)row * TOT + 2 * b];
    float yc = h[(size_t)row * TOT + 2 * b + 1];
    hb[rl][2 * b]     = c * xc - s * yc;   // transpose=True
    hb[rl][2 * b + 1] = s * xc + c * yc;
  }
  __syncthreads();
  int rl = t >> 6, j = t & 63;         // 4 rows x 64 out cols
  int row = row0 + rl;
  float s = ltb[j];
  const float* w = ltw + (size_t)j * TOT;
  #pragma unroll 8
  for (int k = 0; k < TOT; ++k) s = fmaf(hb[rl][k], w[k], s);
  resultT[(size_t)j * KDIM + row] = s;
  termT[(size_t)j * KDIM + row]   = (__bf16)s;
}

// ------------------- Taylor diffusion: partial GEMM ------------------------
// Grid: (KDIM/16) x KSPLIT, one wave per block. Block (m,s) computes the
// 16(M) x 64(N) strip of sum_{k in split s} L[m,k]*term[n,k] and stores raw
// f32 partials (contiguous v8f). 2-stage software pipeline: loads for chunk
// k+32/k+64 are issued before the WMMAs that consume chunk k complete.
__global__ __launch_bounds__(32) void diffuse_partial_kernel(
    const __bf16* __restrict__ Lb,      // 8192 x 8192 row-major bf16
    const __bf16* __restrict__ termIn,  // 64 x 8192 (transposed) bf16
    float* __restrict__ part) {         // KSPLIT x 64 x 8192 f32
  const int lane   = threadIdx.x;       // 0..31, no divergence (EXEC all ones)
  const int m0     = blockIdx.x * 16;
  const int split  = blockIdx.y;
  const int kstart = split * KCHUNK;
  const int mrow   = m0 + (lane & 15);
  const int kbA    = (lane >> 4) * 16;  // A-fragment K sub-offset (contiguous, matches B)
  const int kbB    = (lane >> 4) * 16;  // B-fragment K sub-offset
  const int nlane  = lane & 15;

  const __bf16* arow = Lb + (size_t)mrow * KDIM + kstart + kbA;
  const __bf16* b0p  = termIn + (size_t)(nlane)      * KDIM + kstart + kbB;
  const __bf16* b1p  = termIn + (size_t)(16 + nlane) * KDIM + kstart + kbB;
  const __bf16* b2p  = termIn + (size_t)(32 + nlane) * KDIM + kstart + kbB;
  const __bf16* b3p  = termIn + (size_t)(48 + nlane) * KDIM + kstart + kbB;

  auto loadA = [&](int k) -> v16bf {
    v8bf lo = *(const v8bf*)(arow + k);        // K {0..7}+base
    v8bf hi = *(const v8bf*)(arow + k + 8);    // K {8..15}+base
    return __builtin_shufflevector(lo, hi, 0, 1, 2, 3, 4, 5, 6, 7,
                                   8, 9, 10, 11, 12, 13, 14, 15);
  };

  v8f acc0 = {}, acc1 = {}, acc2 = {}, acc3 = {};

  // Stage 0: chunk k, Stage 1: chunk k+32
  v16bf a0  = loadA(0);
  v16bf b00 = *(const v16bf*)(b0p);
  v16bf b01 = *(const v16bf*)(b1p);
  v16bf b02 = *(const v16bf*)(b2p);
  v16bf b03 = *(const v16bf*)(b3p);
  v16bf a1  = loadA(32);
  v16bf b10 = *(const v16bf*)(b0p + 32);
  v16bf b11 = *(const v16bf*)(b1p + 32);
  v16bf b12 = *(const v16bf*)(b2p + 32);
  v16bf b13 = *(const v16bf*)(b3p + 32);

  for (int k = 0; k < KCHUNK - 64; k += 64) {
    __builtin_prefetch(arow + k + 2048, 0, 1);   // global_prefetch_b8
    // consume stage 0 (chunk k), refill it with chunk k+64
    acc0 = WMMA_BF16(a0, b00, acc0);
    acc1 = WMMA_BF16(a0, b01, acc1);
    acc2 = WMMA_BF16(a0, b02, acc2);
    acc3 = WMMA_BF16(a0, b03, acc3);
    a0  = loadA(k + 64);
    b00 = *(const v16bf*)(b0p + k + 64);
    b01 = *(const v16bf*)(b1p + k + 64);
    b02 = *(const v16bf*)(b2p + k + 64);
    b03 = *(const v16bf*)(b3p + k + 64);
    // consume stage 1 (chunk k+32), refill it with chunk k+96
    acc0 = WMMA_BF16(a1, b10, acc0);
    acc1 = WMMA_BF16(a1, b11, acc1);
    acc2 = WMMA_BF16(a1, b12, acc2);
    acc3 = WMMA_BF16(a1, b13, acc3);
    a1  = loadA(k + 96);
    b10 = *(const v16bf*)(b0p + k + 96);
    b11 = *(const v16bf*)(b1p + k + 96);
    b12 = *(const v16bf*)(b2p + k + 96);
    b13 = *(const v16bf*)(b3p + k + 96);
  }
  // drain both stages
  acc0 = WMMA_BF16(a0, b00, acc0);
  acc1 = WMMA_BF16(a0, b01, acc1);
  acc2 = WMMA_BF16(a0, b02, acc2);
  acc3 = WMMA_BF16(a0, b03, acc3);
  acc0 = WMMA_BF16(a1, b10, acc0);
  acc1 = WMMA_BF16(a1, b11, acc1);
  acc2 = WMMA_BF16(a1, b12, acc2);
  acc3 = WMMA_BF16(a1, b13, acc3);

  // C/D layout: 8 consecutive M per lane at fixed N -> contiguous v8f store
  const int mst = m0 + ((lane >> 4) << 3);
  v8f accs[4] = {acc0, acc1, acc2, acc3};
  #pragma unroll
  for (int tIdx = 0; tIdx < 4; ++tIdx) {
    int n = tIdx * 16 + nlane;
    *(v8f*)(part + ((size_t)split * TOT + n) * KDIM + mst) = accs[tIdx];
  }
}

// ------- reduce K-split partials: t = scale*sum; resultT += t; termOut=bf16(t)
__global__ __launch_bounds__(256) void diffuse_reduce_kernel(
    const float* __restrict__ part,     // KSPLIT x 64 x 8192
    float* __restrict__ resultT,        // 64 x 8192
    __bf16* __restrict__ termOut,       // 64 x 8192
    float scale) {
  size_t base = ((size_t)blockIdx.x * blockDim.x + threadIdx.x) * 4;
  if (base >= (size_t)TOT * KDIM) return;
  float4 s = make_float4(0.f, 0.f, 0.f, 0.f);
  #pragma unroll
  for (int sp = 0; sp < KSPLIT; ++sp) {
    float4 v = *(const float4*)(part + (size_t)sp * TOT * KDIM + base);
    s.x += v.x; s.y += v.y; s.z += v.z; s.w += v.w;
  }
  s.x *= scale; s.y *= scale; s.z *= scale; s.w *= scale;
  float4 r = *(const float4*)(resultT + base);
  r.x += s.x; r.y += s.y; r.z += s.z; r.w += s.w;
  *(float4*)(resultT + base) = r;
  v4bf tb;
  tb[0] = (__bf16)s.x; tb[1] = (__bf16)s.y;
  tb[2] = (__bf16)s.z; tb[3] = (__bf16)s.w;
  *(v4bf*)(termOut + base) = tb;
}

// ---------- rotate(transpose=False) + residual GELU: h += gelu(rot) --------
__global__ void rotate_back_kernel(const float* __restrict__ resultT,
                                   const float* __restrict__ ang,
                                   float* __restrict__ h) {
  int g = blockIdx.x * blockDim.x + threadIdx.x;
  if (g >= NROWS * NBUN) return;
  int row = g >> 5, b = g & 31;
  float a = ang[g];
  float c = cosf(a), s = sinf(a);
  float xc = resultT[(size_t)(2 * b) * KDIM + row];
  float yc = resultT[(size_t)(2 * b + 1) * KDIM + row];
  float xr =  c * xc + s * yc;                 // transpose=False
  float yr = -s * xc + c * yc;
  h[(size_t)row * TOT + 2 * b]     += gelu_exact(xr);
  h[(size_t)row * TOT + 2 * b + 1] += gelu_exact(yr);
}

// ---------------------------------------------------------------------------
extern "C" void kernel_launch(void* const* d_in, const int* in_sizes, int n_in,
                              void* d_out, int out_size, void* d_ws, size_t ws_size,
                              hipStream_t stream) {
  const float* x = (const float*)d_in[0];
  const float* L = (const float*)d_in[1];

  // Param flattening: sorted-key tree order vs insertion order, detected by
  // size of d_in[2] (in_b=64 -> sorted; in_w=8192 -> insertion).
  bool sorted = (in_sizes[2] == TOT);
  const float *in_w, *in_b, *out_w, *out_b;
  int lt_b_off, lt_w_off, pb_off, pw_off;
  if (sorted) {
    in_b = (const float*)d_in[2];  in_w = (const float*)d_in[3];
    lt_b_off = 0; lt_w_off = 1; pb_off = 2; pw_off = 6;
    out_b = (const float*)d_in[44]; out_w = (const float*)d_in[45];
  } else {
    in_w = (const float*)d_in[2];  in_b = (const float*)d_in[3];
    pw_off = 0; pb_off = 4; lt_w_off = 8; lt_b_off = 9;
    out_w = (const float*)d_in[44]; out_b = (const float*)d_in[45];
  }

  // Workspace carve-up (~155 MB total)
  char* ws = (char*)d_ws;
  __bf16* Lb    = (__bf16*)ws;                               // 128 MB
  size_t off = (size_t)KDIM * KDIM * 2;
  float*  h     = (float*)(ws + off);  off += (size_t)NROWS * TOT * 4;    // 2 MB
  float*  phiA  = (float*)(ws + off);  off += (size_t)NROWS * TOT * 4;    // 2 MB
  float*  phiB  = (float*)(ws + off);  off += (size_t)NROWS * TOT * 4;    // 2 MB
  float*  angles= (float*)(ws + off);  off += (size_t)NROWS * NBUN * 4;   // 1 MB
  float*  resT  = (float*)(ws + off);  off += (size_t)TOT * KDIM * 4;     // 2 MB
  __bf16* tA    = (__bf16*)(ws + off); off += (size_t)TOT * KDIM * 2;     // 1 MB
  __bf16* tB    = (__bf16*)(ws + off); off += (size_t)TOT * KDIM * 2;     // 1 MB
  float*  part  = (float*)(ws + off);  // KSPLIT * 64 * 8192 * 4 = 16 MB
  (void)ws_size; (void)n_in; (void)out_size;

  // L -> bf16 (once per call; L2-resident afterwards)
  cvt_bf16_kernel<<<8192, 256, 0, stream>>>(L, Lb, (long)KDIM * KDIM);

  // h = x @ in_w^T + in_b
  gemm_act_kernel<<<(NROWS * TOT + 255) / 256, 256, 0, stream>>>(
      x, in_w, in_b, h, NROWS, D_IN, TOT, 0);

  for (int l = 0; l < 4; ++l) {
    int base = 4 + 10 * l;
    const float* ltb = (const float*)d_in[base + lt_b_off];
    const float* ltw = (const float*)d_in[base + lt_w_off];
    const float* pb0 = (const float*)d_in[base + pb_off + 0];
    const float* pb1 = (const float*)d_in[base + pb_off + 1];
    const float* pb2 = (const float*)d_in[base + pb_off + 2];
    const float* pb3 = (const float*)d_in[base + pb_off + 3];
    const float* pw0 = (const float*)d_in[base + pw_off + 0];
    const float* pw1 = (const float*)d_in[base + pw_off + 1];
    const float* pw2 = (const float*)d_in[base + pw_off + 2];
    const float* pw3 = (const float*)d_in[base + pw_off + 3];

    // phi MLP: 3x GELU(64->64), then linear 64->32
    gemm_act_kernel<<<(NROWS * TOT + 255) / 256, 256, 0, stream>>>(
        h, pw0, pb0, phiA, NROWS, TOT, TOT, 1);
    gemm_act_kernel<<<(NROWS * TOT + 255) / 256, 256, 0, stream>>>(
        phiA, pw1, pb1, phiB, NROWS, TOT, TOT, 1);
    gemm_act_kernel<<<(NROWS * TOT + 255) / 256, 256, 0, stream>>>(
        phiB, pw2, pb2, phiA, NROWS, TOT, TOT, 1);
    gemm_act_kernel<<<(NROWS * NBUN + 255) / 256, 256, 0, stream>>>(
        phiA, pw3, pb3, angles, NROWS, TOT, NBUN, 0);

    // H = rotate_T(h) @ lt_w^T + lt_b  -> resT (f32) and tA (bf16), transposed
    rotate_lt_kernel<<<NROWS / 4, 256, 0, stream>>>(h, angles, ltw, ltb, resT, tA);

    // K=4 Taylor steps: term = (-1/k) L@term; result += term
    __bf16 *ti = tA, *to = tB;
    for (int k = 1; k <= 4; ++k) {
      dim3 grid(KDIM / 16, KSPLIT);
      diffuse_partial_kernel<<<grid, 32, 0, stream>>>(Lb, ti, part);
      diffuse_reduce_kernel<<<(TOT * KDIM / 4 + 255) / 256, 256, 0, stream>>>(
          part, resT, to, -1.0f / (float)k);
      __bf16* tmp = ti; ti = to; to = tmp;
    }

    // h += gelu(rotate_F(result))
    rotate_back_kernel<<<(NROWS * NBUN + 255) / 256, 256, 0, stream>>>(
        resT, angles, h);
  }

  // out = h @ out_w^T + out_b
  gemm_act_kernel<<<(NROWS * TOT + 255) / 256, 256, 0, stream>>>(
      h, out_w, out_b, (float*)d_out, NROWS, TOT, TOT, 0);
}
